// QuantumInspiredRouter_37907381354846
// MI455X (gfx1250) — matrix operand order repaired and checked by reading, not compiled, not run on hardware
//
#include <hip/hip_runtime.h>
#include <math.h>

// ---------------------------------------------------------------------------
// QuantumInspiredRouter for MI455X (gfx1250, wave32, WMMA)
//
//   prep_w1 / prep_w2 : split fp32 weights into hi/lo bf16 PLANES, swizzled
//                       into exact WMMA B-fragment order (direct b128 loads).
//   fused_mlp         : 32 tokens/block. X tile -> LDS hi/lo bf16 planes,
//                       GEMM1 (K=1024,N=1024) via v_wmma_f32_16x16x32_bf16
//                       with 3-product split accumulation (~fp32 accuracy),
//                       bias + exact erf-GELU, H -> LDS planes,
//                       GEMM2 (K=512,N=16 x2 MLPs) -> raw amps/phases.
//   quantum           : per-token softmax/sqrt, 120 Givens rotations,
//                       normalization, top-2, tanh*pi phases.
// ---------------------------------------------------------------------------

#define TOK   16384
#define HDIM  1024
#define HHALF 512

typedef __attribute__((ext_vector_type(16))) __bf16 v16bf;
typedef __attribute__((ext_vector_type(8)))  float  v8f;

union Frag { v16bf v; uint4 q[2]; };

#define WMMA_BF16(A, B, C) \
  __builtin_amdgcn_wmma_f32_16x16x32_bf16(false, (A), false, (B), (short)0, (C), false, false)

// Split fp32 -> (hi bf16, lo bf16 residual), RNE both; returns hi | lo<<16.
__device__ __forceinline__ unsigned int pack_split(float x) {
  unsigned int u  = __float_as_uint(x);
  unsigned int hi = ((u + 0x7FFFu + ((u >> 16) & 1u)) >> 16) & 0xFFFFu;
  float        r  = x - __uint_as_float(hi << 16);
  unsigned int ur = __float_as_uint(r);
  unsigned int lo = ((ur + 0x7FFFu + ((ur >> 16) & 1u)) >> 16) & 0xFFFFu;
  return hi | (lo << 16);
}

// Fragment loads: data pre-arranged so bytes land directly in v16bf layout.
__device__ __forceinline__ v16bf load_frag_lds(const unsigned int* plane, int base) {
  Frag f;
  f.q[0] = *(const uint4*)(plane + base);      // k offsets {0..7}  (8 bf16)
  f.q[1] = *(const uint4*)(plane + base + 8);  // k offsets {16..23}
  return f.v;
}
__device__ __forceinline__ v16bf load_frag_g(const unsigned int* p) {
  Frag f;
  f.q[0] = *(const uint4*)(p);
  f.q[1] = *(const uint4*)(p + 4);
  return f.v;
}

__device__ __forceinline__ float gelu_exact(float x) {
  return 0.5f * x * (1.0f + erff(x * 0.70710678118654752f));
}

// ---------------------------------------------------------------------------
// Weight prep: B fragment (dense 32x16 B: lanes 0-15 hold K 0..7,16..23, lanes
// 16-31 hold K 8..15,24..31; element i = K sub*8+i (i<8), 8+sub*8+i (i>=8)). One u32 output per thread:
// ws index t = ((nt*NKC + kc)*2 + plane)*256 + lane*8 + j,  elements 2j,2j+1.
// ---------------------------------------------------------------------------
__global__ void prep_w1_kernel(const float* __restrict__ w1a, const float* __restrict__ w1p,
                               unsigned int* __restrict__ dst) {
  int t    = blockIdx.x * 256 + threadIdx.x;   // < 1<<20
  int j    = t & 7;
  int lane = (t >> 3) & 31;
  int pl   = (t >> 8) & 1;                     // 0 = hi plane, 1 = lo plane
  int kc   = (t >> 9) & 31;
  int nt   = t >> 14;
  int nl = lane & 15, sub = lane >> 4;
  int col = nt * 16 + nl;                      // 0..1023 : [w1a | w1p]
  int k0  = kc * 32 + (j >> 2) * 16 + sub * 8 + 2 * (j & 3);
  const float* w = (col < HHALF) ? w1a : w1p;
  int c = (col < HHALF) ? col : col - HHALF;
  unsigned int p0 = pack_split(w[k0 * HHALF + c]);
  unsigned int p1 = pack_split(w[(k0 + 1) * HHALF + c]);
  unsigned int b0 = pl ? (p0 >> 16) : (p0 & 0xFFFFu);
  unsigned int b1 = pl ? (p1 >> 16) : (p1 & 0xFFFFu);
  dst[t] = b0 | (b1 << 16);
}

__global__ void prep_w2_kernel(const float* __restrict__ w2a, const float* __restrict__ w2p,
                               unsigned int* __restrict__ dst) {
  int t    = blockIdx.x * 256 + threadIdx.x;   // < 16384
  int j    = t & 7;
  int lane = (t >> 3) & 31;
  int pl   = (t >> 8) & 1;
  int kc   = (t >> 9) & 15;
  int mlp  = t >> 13;
  int nl = lane & 15, sub = lane >> 4;
  int k0  = kc * 32 + (j >> 2) * 16 + sub * 8 + 2 * (j & 3);
  const float* w2 = mlp ? w2p : w2a;
  unsigned int p0 = pack_split(w2[k0 * 16 + nl]);
  unsigned int p1 = pack_split(w2[(k0 + 1) * 16 + nl]);
  unsigned int b0 = pl ? (p0 >> 16) : (p0 & 0xFFFFu);
  unsigned int b1 = pl ? (p1 >> 16) : (p1 & 0xFFFFu);
  dst[t] = b0 | (b1 << 16);
}

// ---------------------------------------------------------------------------
// Fused MLP kernel: 256 threads (8 wave32), 32 tokens per block.
// LDS u32 map: [0,16K) X_hi | [16K,32K) X_lo | [32K,48K) H_hi | [48K,64K) H_lo
// (each plane: 32 rows x 1024 bf16 = 512 u32/row, 64 KB)
// ---------------------------------------------------------------------------
__global__ __launch_bounds__(256, 1)
void fused_mlp_kernel(const float* __restrict__ hs,
                      const float* __restrict__ b1a, const float* __restrict__ b1p,
                      const float* __restrict__ b2a, const float* __restrict__ b2p,
                      const unsigned int* __restrict__ wsw1,
                      const unsigned int* __restrict__ wsw2,
                      float* __restrict__ act) {
  extern __shared__ unsigned int smem[];
  unsigned int* Xhi = smem;
  unsigned int* Xlo = smem + 16384;
  unsigned int* Hhi = smem + 32768;
  unsigned int* Hlo = smem + 49152;
  const int lane = threadIdx.x & 31;
  const int wv   = threadIdx.x >> 5;
  const int nl   = lane & 15;
  const int sub  = lane >> 4;
  const int tok0 = blockIdx.x * 32;

  // ---- Phase 0: X tile [32 x 1024] f32 -> split bf16 planes in LDS ----
  const float4* Xg = (const float4*)(hs + (size_t)tok0 * HDIM);
#pragma unroll 4
  for (int it = 0; it < 32; ++it) {
    int fi   = threadIdx.x + it * 256;   // float4 index; 4 consecutive k, one row
    float4 x = Xg[fi];
    unsigned int s0 = pack_split(x.x), s1 = pack_split(x.y);
    unsigned int s2 = pack_split(x.z), s3 = pack_split(x.w);
    uint2 h, l;
    h.x = (s0 & 0xFFFFu) | ((s1 & 0xFFFFu) << 16);
    h.y = (s2 & 0xFFFFu) | ((s3 & 0xFFFFu) << 16);
    l.x = (s0 >> 16) | ((s1 >> 16) << 16);
    l.y = (s2 >> 16) | ((s3 >> 16) << 16);
    *(uint2*)&Xhi[fi * 2] = h;
    *(uint2*)&Xlo[fi * 2] = l;
  }
  __syncthreads();

  // ---- Phase 1: GEMM1  X[32x1024] @ W1[1024x1024] (+bias, exact GELU) ----
  unsigned short* Hhi16 = (unsigned short*)Hhi;
  unsigned short* Hlo16 = (unsigned short*)Hlo;
  for (int nt = wv * 8; nt < wv * 8 + 8; ++nt) {
    v8f acc0 = {};
    v8f acc1 = {};
    const unsigned int* bw = wsw1 + (size_t)nt * 16384 + lane * 8;  // +kc*512 (hi), +256 (lo)
    for (int kc = 0; kc < 32; ++kc) {
      v16bf bh = load_frag_g(bw + kc * 512);
      v16bf bl = load_frag_g(bw + kc * 512 + 256);
      __builtin_prefetch((const void*)(bw + (kc + 1) * 512), 0, 1);
      int ab = nl * 512 + kc * 16 + sub * 4;          // u32 index in plane
      v16bf ah = load_frag_lds(Xhi, ab);
      v16bf al = load_frag_lds(Xlo, ab);
      acc0 = WMMA_BF16(al, bh, acc0);
      acc0 = WMMA_BF16(ah, bl, acc0);
      acc0 = WMMA_BF16(ah, bh, acc0);
      ah = load_frag_lds(Xhi, ab + 8192);             // rows 16..31
      al = load_frag_lds(Xlo, ab + 8192);
      acc1 = WMMA_BF16(al, bh, acc1);
      acc1 = WMMA_BF16(ah, bl, acc1);
      acc1 = WMMA_BF16(ah, bh, acc1);
    }
    int col  = nt * 16 + nl;
    float bv = (col < HHALF) ? b1a[col] : b1p[col - HHALF];
#pragma unroll
    for (int r = 0; r < 8; ++r) {
      // D layout: lane = column, VGPR r = row r (+8 for lanes 16-31)
      unsigned int p0 = pack_split(gelu_exact(acc0[r] + bv));
      unsigned int p1 = pack_split(gelu_exact(acc1[r] + bv));
      int i0 = (r + sub * 8) * 1024 + col;
      int i1 = i0 + 16 * 1024;
      Hhi16[i0] = (unsigned short)(p0 & 0xFFFFu);
      Hlo16[i0] = (unsigned short)(p0 >> 16);
      Hhi16[i1] = (unsigned short)(p1 & 0xFFFFu);
      Hlo16[i1] = (unsigned short)(p1 >> 16);
    }
  }
  __syncthreads();

  // ---- Phase 2: GEMM2  H[32x512] @ W2[512x16] per MLP (waves 0..3) ----
  if (wv < 4) {
    int m = wv & 1, mlp = wv >> 1;
    v8f acc = {};
    const unsigned int* bw = wsw2 + (size_t)mlp * 8192 + lane * 8;  // +kc*512 (hi), +256 (lo)
    for (int kc = 0; kc < 16; ++kc) {
      v16bf bh = load_frag_g(bw + kc * 512);
      v16bf bl = load_frag_g(bw + kc * 512 + 256);
      int ab = (m * 16 + nl) * 512 + mlp * 256 + kc * 16 + sub * 4;
      v16bf ah = load_frag_lds(Hhi, ab);
      v16bf al = load_frag_lds(Hlo, ab);
      acc = WMMA_BF16(al, bh, acc);
      acc = WMMA_BF16(ah, bl, acc);
      acc = WMMA_BF16(ah, bh, acc);
    }
    float bv = (mlp ? b2p : b2a)[nl];
#pragma unroll
    for (int r = 0; r < 8; ++r) {
      int tg = tok0 + m * 16 + r + sub * 8;
      act[(size_t)tg * 32 + mlp * 16 + nl] = acc[r] + bv;  // [amps(16) | phases(16)]
    }
  }
}

// ---------------------------------------------------------------------------
// Quantum routing elementwise kernel: one thread per token.
// ---------------------------------------------------------------------------
__global__ void quantum_kernel(const float* __restrict__ act, const float* __restrict__ ent,
                               float* __restrict__ top_p, int* __restrict__ top_i,
                               float* __restrict__ dec, float* __restrict__ phs,
                               float* __restrict__ prb) {
  __shared__ float cs_s[120], sn_s[120];
  int t = threadIdx.x;
  if (t < 120) {
    int idx = 0, pi = 0, pj = 1;
    for (int a = 0; a < 16; ++a)
      for (int b = a + 1; b < 16; ++b) {
        if (idx == t) { pi = a; pj = b; }
        ++idx;
      }
    float ang = ent[pi * 16 + pj] * 0.5f;   // ENT_STRENGTH = 0.5
    cs_s[t] = cosf(ang);
    sn_s[t] = sinf(ang);
  }
  __syncthreads();

  int tok = blockIdx.x * 256 + threadIdx.x;
  const float* a = act + (size_t)tok * 32;

  float amp[16];
#pragma unroll
  for (int e = 0; e < 16; ++e) amp[e] = fabsf(a[e]);
#pragma unroll
  for (int e = 0; e < 16; ++e)
    phs[tok * 16 + e] = tanhf(a[16 + e]) * 3.14159265358979323846f;

  // sqrt(softmax(|amps|))
  float mx = amp[0];
#pragma unroll
  for (int e = 1; e < 16; ++e) mx = fmaxf(mx, amp[e]);
  float s = 0.f;
#pragma unroll
  for (int e = 0; e < 16; ++e) { amp[e] = expf(amp[e] - mx); s += amp[e]; }
  float inv = 1.f / s;
#pragma unroll
  for (int e = 0; e < 16; ++e) amp[e] = sqrtf(amp[e] * inv);

  // 120 sequential Givens rotations (reference scan order)
  int idx = 0;
#pragma unroll
  for (int i = 0; i < 16; ++i) {
#pragma unroll
    for (int j = i + 1; j < 16; ++j) {
      float c = cs_s[idx], sn = sn_s[idx];
      float ai = amp[i], aj = amp[j];
      amp[i] = c * ai - sn * aj;
      amp[j] = sn * ai + c * aj;
      ++idx;
    }
  }

  // coherence = exp(-0/10) = 1.0
  float pr[16];
  float ss = 0.f;
#pragma unroll
  for (int e = 0; e < 16; ++e) {
    dec[tok * 16 + e] = amp[e];
    pr[e] = amp[e] * amp[e];
    ss += fabsf(pr[e]);
  }
  float den = fmaxf(ss, 1e-12f);
  inv = 1.f / den;
#pragma unroll
  for (int e = 0; e < 16; ++e) { pr[e] *= inv; prb[tok * 16 + e] = pr[e]; }

  // top-2, lowest-index tie-breaking (jax top_k semantics)
  int i0 = 0; float v0 = pr[0];
#pragma unroll
  for (int e = 1; e < 16; ++e) if (pr[e] > v0) { v0 = pr[e]; i0 = e; }
  int i1 = (i0 == 0) ? 1 : 0; float v1 = pr[i1];
#pragma unroll
  for (int e = 0; e < 16; ++e) if (e != i0 && pr[e] > v1) { v1 = pr[e]; i1 = e; }

  float d2 = fmaxf(fabsf(v0) + fabsf(v1), 1e-12f);
  top_p[tok * 2]     = v0 / d2;
  top_p[tok * 2 + 1] = v1 / d2;
  top_i[tok * 2]     = i0;
  top_i[tok * 2 + 1] = i1;
}

// ---------------------------------------------------------------------------
extern "C" void kernel_launch(void* const* d_in, const int* in_sizes, int n_in,
                              void* d_out, int out_size, void* d_ws, size_t ws_size,
                              hipStream_t stream) {
  const float* hs  = (const float*)d_in[0];
  const float* w1a = (const float*)d_in[1];
  const float* b1a = (const float*)d_in[2];
  const float* w2a = (const float*)d_in[3];
  const float* b2a = (const float*)d_in[4];
  const float* w1p = (const float*)d_in[5];
  const float* b1p = (const float*)d_in[6];
  const float* w2p = (const float*)d_in[7];
  const float* b2p = (const float*)d_in[8];
  const float* ent = (const float*)d_in[9];

  // Workspace (u32 units): w1 planes 1M, w2 planes 16K, act 512K f32
  unsigned int* wsw1 = (unsigned int*)d_ws;
  unsigned int* wsw2 = wsw1 + (1u << 20);
  float*        act  = (float*)(wsw2 + 16384);

  // Output (tuple flattened): top_p[32768] f32, top_i[32768] i32,
  // decoherent[262144], raw_phases[262144], probs[262144]
  float* out_f = (float*)d_out;
  float* top_p = out_f;
  int*   top_i = (int*)(out_f + 32768);
  float* dec   = out_f + 65536;
  float* phs   = out_f + 65536 + 262144;
  float* prb   = out_f + 65536 + 2 * 262144;

  prep_w1_kernel<<<4096, 256, 0, stream>>>(w1a, w1p, wsw1);
  prep_w2_kernel<<<64, 256, 0, stream>>>(w2a, w2p, wsw2);
  fused_mlp_kernel<<<TOK / 32, 256, 262144, stream>>>(hs, b1a, b1p, b2a, b2p, wsw1, wsw2, act);
  quantum_kernel<<<TOK / 256, 256, 0, stream>>>(act, ent, top_p, top_i, dec, phs, prb);
}
